// Model_60129542144601
// MI455X (gfx1250) — matrix operand order repaired and checked by reading, not compiled, not run on hardware
//
#include <hip/hip_runtime.h>
#include <hip/hip_bf16.h>

typedef _Float16 f16;
typedef f16   v8h  __attribute__((ext_vector_type(8)));
typedef f16   v16h __attribute__((ext_vector_type(16)));
typedef float v8f  __attribute__((ext_vector_type(8)));
typedef unsigned int u32x4 __attribute__((ext_vector_type(4)));
typedef int i32x8 __attribute__((ext_vector_type(8)));
typedef int i32x4 __attribute__((ext_vector_type(4)));

#define B_  8
#define S_  1024
#define D_  1024
#define H_  16
#define DH_ 64

__device__ __forceinline__ v8f vzero8() {
  v8f z;
#pragma unroll
  for (int i = 0; i < 8; ++i) z[i] = 0.f;
  return z;
}

__device__ __forceinline__ v16h cat16(v8h lo, v8h hi) {
  return __builtin_shufflevector(lo, hi, 0,1,2,3,4,5,6,7,8,9,10,11,12,13,14,15);
}

// A-fragment (16x32 f16, ISA 7.12.2): rowbase -> K=0 of this lane's M-row.
// lane<16: K = {0..7, 16..23}; lane>=16: K = {8..15, 24..31}
__device__ __forceinline__ v16h ldA(const f16* rowbase, int lane) {
  const f16* p = rowbase + ((lane & 16) ? 8 : 0);
  return cat16(*(const v8h*)p, *(const v8h*)(p + 16));
}

// B-fragment (32x16 f16): colbase -> K=0 of this lane's N-column (K contiguous).
__device__ __forceinline__ v16h ldB(const f16* colbase, int lane) {
  const f16* p = colbase + ((lane & 16) ? 8 : 0);
  return cat16(*(const v8h*)p, *(const v8h*)(p + 16));
}

__device__ __forceinline__ v8f wmma_f16(v16h a, v16h b, v8f c) {
  return __builtin_amdgcn_wmma_f32_16x16x32_f16(false, a, false, b, (short)0, c, false, false);
}

// ---------------------------------------------------------------------------
// TDM: DMA a 2D f16 tile (tile1 rows x tile0 elems, row stride stride0 elems)
// from global into LDS with 4-DWORD padding every 16 DWORDs (row stride 32->40
// halves, matching sTile[.][40]).  D# packed per CDNA5 ISA.
// 6-arg builtin form (amdgpu-toolchain / clang-23).
// ---------------------------------------------------------------------------
__device__ __forceinline__ void tdm_load_2d(unsigned lds_addr, const f16* gptr,
                                            unsigned tile0, unsigned tile1,
                                            unsigned stride0) {
  unsigned long long ga = (unsigned long long)(size_t)gptr;
  u32x4 g0;
  g0[0] = 1u;                                         // count=1, user mode
  g0[1] = lds_addr;                                   // lds_addr [63:32]
  g0[2] = (unsigned)(ga & 0xffffffffu);               // global_addr lo
  g0[3] = (unsigned)((ga >> 32) & 0x01ffffffu) | (2u << 30);  // addr hi | type=2
  i32x8 g1;
  // data_size=2B (code 1), pad_enable, pad_interval=16DW (code 3), pad_amount=4DW (code 3)
  g1[0] = (int)((1u << 16) | (1u << 20) | (3u << 22) | (3u << 25));
  g1[1] = (int)((tile0 & 0xffffu) << 16);             // tensor_dim0 = tile0
  g1[2] = (int)((tile0 >> 16) | ((tile1 & 0xffffu) << 16));   // tensor_dim1 = tile1
  g1[3] = (int)((tile1 >> 16) | (tile0 << 16));       // tile_dim0
  g1[4] = (int)(tile1 & 0xffffu);                     // tile_dim1 (tile_dim2 = 0)
  g1[5] = (int)stride0;                               // tensor_dim0_stride lo
  g1[6] = 0;
  g1[7] = 0;
  i32x4 z4;
  z4[0] = 0; z4[1] = 0; z4[2] = 0; z4[3] = 0;
  i32x8 z8;
#pragma unroll
  for (int i = 0; i < 8; ++i) z8[i] = 0;
  __builtin_amdgcn_tensor_load_to_lds(g0, g1, z4, z4, z8, 0);
}

// ---------------------------------------------------------------------------
// Prep: f32 -> f16 elementwise convert (x -> x16)
// ---------------------------------------------------------------------------
__global__ __launch_bounds__(256) void k_cvt(const float* __restrict__ src,
                                             f16* __restrict__ dst) {
  size_t i = ((size_t)blockIdx.x * 256 + threadIdx.x) * 8;
#pragma unroll
  for (int j = 0; j < 8; ++j) dst[i + j] = (f16)src[i + j];
}

// Prep: W (K x N f32) -> WT (N x K f16) via LDS tile transpose
__global__ __launch_bounds__(256) void k_cvtT(const float* __restrict__ W,
                                              f16* __restrict__ WT,
                                              int K, int N) {
  __shared__ f16 s[64][72];
  const int n0 = blockIdx.x * 64, k0 = blockIdx.y * 64;
  const int t = threadIdx.x;
  const int r = t >> 2, c0 = (t & 3) * 16;
  const float* src = W + (size_t)(k0 + r) * N + n0 + c0;
#pragma unroll
  for (int i = 0; i < 16; ++i) s[c0 + i][r] = (f16)src[i];   // s[n][k]
  __syncthreads();
  f16* dst = WT + (size_t)(n0 + r) * K + k0 + c0;
#pragma unroll
  for (int i = 0; i < 16; ++i) dst[i] = s[r][c0 + i];
}

// ---------------------------------------------------------------------------
// Kernel 1: QKV projection, TDM-staged + double-buffered.
// C(8192x3072) = x16(8192x1024) @ WqkvT^T + bqkv
// q scaled by DH^-0.5 -> (B,H,S,DH); k -> (B,H,S,DH); v transposed -> (B,H,DH,S)
// ---------------------------------------------------------------------------
__global__ __launch_bounds__(256) void k_qkv(const f16* __restrict__ x16,
                                             const f16* __restrict__ WqkvT,
                                             const float* __restrict__ bqkv,
                                             f16* __restrict__ q16,
                                             f16* __restrict__ k16,
                                             f16* __restrict__ vT16) {
  __shared__ f16 sA[2][128][40];
  __shared__ f16 sB[2][128][40];
  const int n0 = blockIdx.x * 128;
  const int m0 = blockIdx.y * 128;
  const int t = threadIdx.x, lane = t & 31, wave = t >> 5;

  v8f acc[8];
#pragma unroll
  for (int j = 0; j < 8; ++j) acc[j] = vzero8();

  if (wave == 0) {
    tdm_load_2d((unsigned)(size_t)&sA[0][0][0], x16 + (size_t)m0 * D_, 32, 128, D_);
    tdm_load_2d((unsigned)(size_t)&sB[0][0][0], WqkvT + (size_t)n0 * D_, 32, 128, D_);
  }
  for (int kk = 0; kk < D_; kk += 32) {
    const int buf = (kk >> 5) & 1;
    if (wave == 0) {
      if (kk + 32 < D_) {
        tdm_load_2d((unsigned)(size_t)&sA[buf ^ 1][0][0],
                    x16 + (size_t)m0 * D_ + kk + 32, 32, 128, D_);
        tdm_load_2d((unsigned)(size_t)&sB[buf ^ 1][0][0],
                    WqkvT + (size_t)n0 * D_ + kk + 32, 32, 128, D_);
        __builtin_amdgcn_s_wait_tensorcnt(2);   // current pair complete
      } else {
        __builtin_amdgcn_s_wait_tensorcnt(0);
      }
    }
    __syncthreads();
    v16h a = ldA(&sA[buf][wave * 16 + (lane & 15)][0], lane);
#pragma unroll
    for (int j = 0; j < 8; ++j) {
      v16h b = ldB(&sB[buf][j * 16 + (lane & 15)][0], lane);
      acc[j] = wmma_f16(a, b, acc[j]);
    }
    __syncthreads();
  }

  const int mrow = m0 + wave * 16 + ((lane & 16) ? 8 : 0);
#pragma unroll
  for (int j = 0; j < 8; ++j) {
    int n = n0 + j * 16 + (lane & 15);
    float bias = bqkv[n];
    int tsel = n >> 10;          // 0:q 1:k 2:v (uniform per wave)
    int rem = n & 1023;
    int h = rem >> 6, d = rem & 63;
#pragma unroll
    for (int r = 0; r < 8; ++r) {
      int m = mrow + r;
      int b = m >> 10, s = m & 1023;
      float val = acc[j][r] + bias;
      size_t idx = ((size_t)(b * H_ + h) * S_ + s) * DH_ + d;
      if (tsel == 0)      q16[idx] = (f16)(val * 0.125f);   // fold DH^-0.5
      else if (tsel == 1) k16[idx] = (f16)val;
      else vT16[((size_t)(b * H_ + h) * DH_ + d) * S_ + s] = (f16)val;
    }
  }
}

// ---------------------------------------------------------------------------
// Kernel 2: flash-style attention per (b,h); online softmax, key blocks of 32.
// ---------------------------------------------------------------------------
__global__ __launch_bounds__(256) void k_attn(const f16* __restrict__ q16,
                                              const f16* __restrict__ k16,
                                              const f16* __restrict__ vT16,
                                              const float* __restrict__ mask,
                                              f16* __restrict__ ctx16) {
  __shared__ f16 sP[8][16][40];
  const int q0 = blockIdx.x * 128;
  const int bh = blockIdx.y;
  const int b = bh >> 4, h = bh & 15;
  const int t = threadIdx.x, lane = t & 31, wave = t >> 5;

  const f16* qbase = q16 + ((size_t)bh * S_ + q0 + wave * 16) * DH_;
  const f16* kbase = k16 + (size_t)bh * S_ * DH_;
  const f16* vbase = vT16 + (size_t)bh * DH_ * S_;

  v16h qa0 = ldA(qbase + (size_t)(lane & 15) * DH_ + 0, lane);
  v16h qa1 = ldA(qbase + (size_t)(lane & 15) * DH_ + 32, lane);

  float mrun[8], lrun[8];
  v8f acc[4];
#pragma unroll
  for (int r = 0; r < 8; ++r) { mrun[r] = -3.0e38f; lrun[r] = 0.f; }
#pragma unroll
  for (int j = 0; j < 4; ++j) acc[j] = vzero8();

  const float* mbase = mask +
      ((size_t)b * S_ + q0 + wave * 16 + ((lane & 16) ? 8 : 0)) * S_ + (lane & 15);

  for (int kb = 0; kb < S_; kb += 32) {
    if (kb + 32 < S_) {   // prefetch next key/value block (global_prefetch)
      __builtin_prefetch(kbase + (size_t)(kb + 32 + (lane & 15)) * DH_, 0, 0);
      __builtin_prefetch(vbase + (size_t)(lane & 15) * S_ + kb + 32, 0, 0);
    }
    v8f sc0 = vzero8(), sc1 = vzero8();
    {
      const f16* kr0 = kbase + (size_t)(kb + (lane & 15)) * DH_;
      const f16* kr1 = kbase + (size_t)(kb + 16 + (lane & 15)) * DH_;
      v16h b00 = ldB(kr0 + 0, lane);
      v16h b01 = ldB(kr0 + 32, lane);
      v16h b10 = ldB(kr1 + 0, lane);
      v16h b11 = ldB(kr1 + 32, lane);
      sc0 = wmma_f16(qa0, b00, sc0);
      sc0 = wmma_f16(qa1, b01, sc0);
      sc1 = wmma_f16(qa0, b10, sc1);
      sc1 = wmma_f16(qa1, b11, sc1);
    }
    float p0[8], p1[8], alpha[8];
#pragma unroll
    for (int r = 0; r < 8; ++r) {
      float s0 = sc0[r] + mbase[(size_t)r * S_ + kb] * -1.0e30f;
      float s1 = sc1[r] + mbase[(size_t)r * S_ + kb + 16] * -1.0e30f;
      float mx = fmaxf(s0, s1);
      mx = fmaxf(mx, __shfl_xor(mx, 1));
      mx = fmaxf(mx, __shfl_xor(mx, 2));
      mx = fmaxf(mx, __shfl_xor(mx, 4));
      mx = fmaxf(mx, __shfl_xor(mx, 8));
      float mnew = fmaxf(mrun[r], mx);
      float a = __expf(mrun[r] - mnew);
      mrun[r] = mnew;
      alpha[r] = a;
      p0[r] = __expf(s0 - mnew);
      p1[r] = __expf(s1 - mnew);
      float rs = p0[r] + p1[r];
      rs += __shfl_xor(rs, 1);
      rs += __shfl_xor(rs, 2);
      rs += __shfl_xor(rs, 4);
      rs += __shfl_xor(rs, 8);
      lrun[r] = lrun[r] * a + rs;
    }
#pragma unroll
    for (int j = 0; j < 4; ++j)
#pragma unroll
      for (int r = 0; r < 8; ++r) acc[j][r] *= alpha[r];

    {
      int rr = (lane & 16) ? 8 : 0;
      int c = lane & 15;
#pragma unroll
      for (int r = 0; r < 8; ++r) {
        sP[wave][r + rr][c]      = (f16)p0[r];
        sP[wave][r + rr][c + 16] = (f16)p1[r];
      }
    }
    __syncthreads();
    v16h pa = ldA(&sP[wave][lane & 15][0], lane);
#pragma unroll
    for (int j = 0; j < 4; ++j) {
      const f16* vr = vbase + (size_t)(j * 16 + (lane & 15)) * S_ + kb;
      v16h vb = ldB(vr, lane);
      acc[j] = wmma_f16(pa, vb, acc[j]);
    }
    __syncthreads();
  }

  const int row = q0 + wave * 16 + ((lane & 16) ? 8 : 0);
  f16* cbase = ctx16 + ((size_t)b * S_ + row) * D_ + h * DH_ + (lane & 15);
#pragma unroll
  for (int r = 0; r < 8; ++r) {
    float inv = 1.0f / lrun[r];
#pragma unroll
    for (int j = 0; j < 4; ++j)
      cbase[(size_t)r * D_ + j * 16] = (f16)(acc[j][r] * inv);
  }
}

// ---------------------------------------------------------------------------
// Kernel 3: out = x + ctx16 @ WoutT^T + bout  (TDM-staged, double-buffered)
// ---------------------------------------------------------------------------
__global__ __launch_bounds__(256) void k_out(const f16* __restrict__ ctx16,
                                             const f16* __restrict__ WoutT,
                                             const float* __restrict__ bout,
                                             const float* __restrict__ x,
                                             float* __restrict__ out) {
  __shared__ f16 sA[2][128][40];
  __shared__ f16 sB[2][128][40];
  const int n0 = blockIdx.x * 128;
  const int m0 = blockIdx.y * 128;
  const int t = threadIdx.x, lane = t & 31, wave = t >> 5;

  v8f acc[8];
#pragma unroll
  for (int j = 0; j < 8; ++j) acc[j] = vzero8();

  if (wave == 0) {
    tdm_load_2d((unsigned)(size_t)&sA[0][0][0], ctx16 + (size_t)m0 * D_, 32, 128, D_);
    tdm_load_2d((unsigned)(size_t)&sB[0][0][0], WoutT + (size_t)n0 * D_, 32, 128, D_);
  }
  for (int kk = 0; kk < D_; kk += 32) {
    const int buf = (kk >> 5) & 1;
    if (wave == 0) {
      if (kk + 32 < D_) {
        tdm_load_2d((unsigned)(size_t)&sA[buf ^ 1][0][0],
                    ctx16 + (size_t)m0 * D_ + kk + 32, 32, 128, D_);
        tdm_load_2d((unsigned)(size_t)&sB[buf ^ 1][0][0],
                    WoutT + (size_t)n0 * D_ + kk + 32, 32, 128, D_);
        __builtin_amdgcn_s_wait_tensorcnt(2);
      } else {
        __builtin_amdgcn_s_wait_tensorcnt(0);
      }
    }
    __syncthreads();
    v16h a = ldA(&sA[buf][wave * 16 + (lane & 15)][0], lane);
#pragma unroll
    for (int j = 0; j < 8; ++j) {
      v16h b = ldB(&sB[buf][j * 16 + (lane & 15)][0], lane);
      acc[j] = wmma_f16(a, b, acc[j]);
    }
    __syncthreads();
  }

  const int mrow = m0 + wave * 16 + ((lane & 16) ? 8 : 0);
#pragma unroll
  for (int j = 0; j < 8; ++j) {
    int n = n0 + j * 16 + (lane & 15);
    float bias = bout[n];
#pragma unroll
    for (int r = 0; r < 8; ++r) {
      int m = mrow + r;
      out[(size_t)m * D_ + n] = x[(size_t)m * D_ + n] + acc[j][r] + bias;
    }
  }
}

// ---------------------------------------------------------------------------
extern "C" void kernel_launch(void* const* d_in, const int* in_sizes, int n_in,
                              void* d_out, int out_size, void* d_ws, size_t ws_size,
                              hipStream_t stream) {
  (void)in_sizes; (void)n_in; (void)out_size; (void)ws_size;
  const float* x    = (const float*)d_in[0];
  const float* mask = (const float*)d_in[1];
  const float* Wqkv = (const float*)d_in[2];
  const float* bqkv = (const float*)d_in[3];
  const float* Wout = (const float*)d_in[4];
  const float* bout = (const float*)d_in[5];
  float* out = (float*)d_out;

  const size_t MD = (size_t)B_ * S_ * D_;        // 8Mi
  f16* x16    = (f16*)d_ws;                      // 16 MB
  f16* WqkvT  = x16 + MD;                        //  6 MB
  f16* WoutT  = WqkvT + (size_t)3 * D_ * D_;     //  2 MB
  f16* q16    = WoutT + (size_t)D_ * D_;         // 16 MB
  f16* k16    = q16 + MD;                        // 16 MB
  f16* vT16   = k16 + MD;                        // 16 MB
  f16* ctx16  = vT16 + MD;                       // 16 MB   (total 88 MB)

  k_cvt<<<dim3((unsigned)(MD / (256 * 8))), 256, 0, stream>>>(x, x16);
  k_cvtT<<<dim3((3 * D_) / 64, D_ / 64), 256, 0, stream>>>(Wqkv, WqkvT, D_, 3 * D_);
  k_cvtT<<<dim3(D_ / 64, D_ / 64), 256, 0, stream>>>(Wout, WoutT, D_, D_);

  k_qkv<<<dim3((3 * D_) / 128, (B_ * S_) / 128), 256, 0, stream>>>(
      x16, WqkvT, bqkv, q16, k16, vT16);
  k_attn<<<dim3(S_ / 128, B_ * H_), 256, 0, stream>>>(
      q16, k16, vT16, mask, ctx16);
  k_out<<<dim3(D_ / 128, (B_ * S_) / 128), 256, 0, stream>>>(
      ctx16, WoutT, bout, x, out);
}
